// Decoder_88665304859246
// MI455X (gfx1250) — matrix operand order repaired and checked by reading, not compiled, not run on hardware
//
#include <hip/hip_runtime.h>

// ---------------- problem constants ----------------
#define NM_   80
#define PRE_  256
#define ARNN_ 1024
#define AD_   128
#define ENC_  512
#define NF_   32
#define KS_   31
#define DRNN_ 1024
#define B_    32
#define S_    512
#define T_    600

#define K1DIM (PRE_ + ENC_ + ARNN_)    // 1792
#define K2DIM (ARNN_ + ENC_ + DRNN_)   // 2560
#define KT1   (K1DIM / 32)             // 56 (7 chunks of 8)
#define KT2   (K2DIM / 32)             // 80 (10 chunks of 8)
#define NGATE 4096
#define NTILES (NGATE / 16)            // 256

#define GRID_ 64
#define TPB_  256
#define CHUNK_ 8                        // k-tiles per LDS staging chunk
#define AFRAG_ (CHUNK_ * 2 * 32 * 16)   // __bf16 elems per staging buffer (8192)

// ---------------- WMMA types ----------------
typedef __attribute__((ext_vector_type(16))) __bf16 v16bf;
typedef __attribute__((ext_vector_type(8)))  float  v8f;

__device__ __forceinline__ __bf16 f2bf(float x) {   // manual RNE (setup kernel only)
    unsigned u = __float_as_uint(x);
    unsigned r = (u + 0x7fffu + ((u >> 16) & 1u)) >> 16;
    unsigned short s = (unsigned short)r;
    return __builtin_bit_cast(__bf16, s);
}

__device__ __forceinline__ float sigm(float x) { return 1.0f / (1.0f + expf(-x)); }

// ---------------- grid-wide barrier (persistent kernel) ----------------
__device__ __forceinline__ void grid_barrier(unsigned* cnt, unsigned target) {
    __threadfence();
    __syncthreads();
    if (threadIdx.x == 0) {
        __hip_atomic_fetch_add(cnt, 1u, __ATOMIC_ACQ_REL, __HIP_MEMORY_SCOPE_AGENT);
        while (__hip_atomic_load(cnt, __ATOMIC_ACQUIRE, __HIP_MEMORY_SCOPE_AGENT) < target) {
            __builtin_amdgcn_s_sleep(2);
        }
    }
    __syncthreads();
}

// Gather one 16x32 A-fragment row-slice for this lane: one contiguous 16-float
// run (branch-free, vector loads + native bf16 converts).
__device__ __forceinline__ v16bf gather_frag(const float* __restrict__ sp, int khalf) {
    float4 r0 = *(const float4*)(sp + khalf);
    float4 r1 = *(const float4*)(sp + khalf + 4);
    float4 r2 = *(const float4*)(sp + khalf + 8);
    float4 r3 = *(const float4*)(sp + khalf + 12);
    v16bf a;
    a[0]  = (__bf16)r0.x; a[1]  = (__bf16)r0.y; a[2]  = (__bf16)r0.z; a[3]  = (__bf16)r0.w;
    a[4]  = (__bf16)r1.x; a[5]  = (__bf16)r1.y; a[6]  = (__bf16)r1.z; a[7]  = (__bf16)r1.w;
    a[8]  = (__bf16)r2.x; a[9]  = (__bf16)r2.y; a[10] = (__bf16)r2.z; a[11] = (__bf16)r2.w;
    a[12] = (__bf16)r3.x; a[13] = (__bf16)r3.y; a[14] = (__bf16)r3.z; a[15] = (__bf16)r3.w;
    return a;
}

// All 8 waves of a GEMM block cooperate: wave `wid` computes columns of
// j = jbase+wid for all 4 gates; A fragments are staged chunk-wise through a
// double-buffered LDS tile shared by the whole block.
// A source = 3 row-major segments with k-tile boundaries e0, e1 (in k-tiles).
__device__ __forceinline__ void lstm_gates_gemm_block(
    int wid, int jbase, int lane, const v16bf* __restrict__ Wp, int KT,
    const float* __restrict__ p0, int st0, int e0,
    const float* __restrict__ p1, int st1, int e1,
    const float* __restrict__ p2, int st2,
    const float* __restrict__ bi, const float* __restrict__ bh,
    float* __restrict__ cstate, float* __restrict__ hout,
    __bf16* __restrict__ alds)   // shared, 2 * AFRAG_ elems
{
    const int j = jbase + wid;
    v8f z = {0.f, 0.f, 0.f, 0.f, 0.f, 0.f, 0.f, 0.f};
    v8f acc[4][2];
#pragma unroll
    for (int g = 0; g < 4; ++g) { acc[g][0] = z; acc[g][1] = z; }

    const int mrow  = lane & 15;
    const int khalf = ((lane >> 4) & 1) << 4;
    const int nchunk = KT >> 3;

    // gather for chunk `ch` into buffer buf: wave wid owns k-tile ch*8+wid
    auto gather_chunk = [&](int ch, int buf) {
        const int kt = (ch << 3) + wid;
        const float* base; int stride, rel;
        if (kt < e0)      { base = p0; stride = st0; rel = kt; }
        else if (kt < e1) { base = p1; stride = st1; rel = kt - e0; }
        else              { base = p2; stride = st2; rel = kt - e1; }
        const float* sp = base + mrow * stride + rel * 32;
        __bf16* dst = alds + buf * AFRAG_;
        *(v16bf*)&dst[((wid * 2 + 0) * 32 + lane) * 16] = gather_frag(sp, khalf);
        *(v16bf*)&dst[((wid * 2 + 1) * 32 + lane) * 16] = gather_frag(sp + 16 * stride, khalf);
    };

    gather_chunk(0, 0);
    __syncthreads();
    for (int ch = 0; ch < nchunk; ++ch) {
        if (ch + 1 < nchunk) {
            gather_chunk(ch + 1, (ch + 1) & 1);
#pragma unroll
            for (int g = 0; g < 4; ++g)  // hint next weight chunk (L2-resident)
                __builtin_prefetch(
                    (const void*)&Wp[(((g * 64 + j) * KT) + ((ch + 1) << 3) + wid) * 32 + lane],
                    0, 3);
        }
        const __bf16* src = alds + (ch & 1) * AFRAG_;
#pragma unroll 2
        for (int c = 0; c < CHUNK_; ++c) {
            const int kt = (ch << 3) + c;
            v16bf a0 = *(const v16bf*)&src[((c * 2 + 0) * 32 + lane) * 16];
            v16bf a1 = *(const v16bf*)&src[((c * 2 + 1) * 32 + lane) * 16];
#pragma unroll
            for (int g = 0; g < 4; ++g) {
                v16bf b = Wp[(((g * 64 + j) * KT) + kt) * 32 + lane];
                acc[g][0] = __builtin_amdgcn_wmma_f32_16x16x32_bf16(
                    false, a0, false, b, (short)0, acc[g][0], false, false);
                acc[g][1] = __builtin_amdgcn_wmma_f32_16x16x32_bf16(
                    false, a1, false, b, (short)0, acc[g][1], false, false);
            }
        }
        __syncthreads();
    }

    // pointwise LSTM: lane owns column col, rows r+rofs per C-layout
    const int col  = j * 16 + (lane & 15);
    const int rofs = ((lane >> 4) & 1) << 3;
    const float bgi = bi[0 * 1024 + col] + bh[0 * 1024 + col];
    const float bgf = bi[1 * 1024 + col] + bh[1 * 1024 + col];
    const float bgg = bi[2 * 1024 + col] + bh[2 * 1024 + col];
    const float bgo = bi[3 * 1024 + col] + bh[3 * 1024 + col];
#pragma unroll
    for (int m = 0; m < 2; ++m) {
#pragma unroll
        for (int r = 0; r < 8; ++r) {
            const int row = m * 16 + r + rofs;
            const float gi = sigm(acc[0][m][r] + bgi);
            const float gf = sigm(acc[1][m][r] + bgf);
            const float gg = tanhf(acc[2][m][r] + bgg);
            const float go = sigm(acc[3][m][r] + bgo);
            const int idx = row * 1024 + col;
            const float cn = gf * cstate[idx] + gi * gg;
            cstate[idx] = cn;
            hout[idx]   = go * tanhf(cn);
        }
    }
}

__device__ __forceinline__ void mel_stop_out(
    int o, int tm, const float* __restrict__ dhb, const float* __restrict__ ctx,
    const float* __restrict__ w_mel, const float* __restrict__ b_mel,
    const float* __restrict__ w_stop, const float* __restrict__ b_stop,
    float* __restrict__ out_mel, float* __restrict__ out_stop)
{
    if (o < B_ * NM_) {
        const int b = o / NM_, m = o % NM_;
        float acc = b_mel[m];
        const float* wr = w_mel + m * (DRNN_ + ENC_);
        for (int k = 0; k < DRNN_; ++k) acc += wr[k] * dhb[b * DRNN_ + k];
        for (int k = 0; k < ENC_;  ++k) acc += wr[DRNN_ + k] * ctx[b * ENC_ + k];
        out_mel[(b * T_ + tm) * NM_ + m] = acc;
    } else {
        const int b = o - B_ * NM_;
        float acc = b_stop[0];
        for (int k = 0; k < DRNN_; ++k) acc += w_stop[k] * dhb[b * DRNN_ + k];
        for (int k = 0; k < ENC_;  ++k) acc += w_stop[DRNN_ + k] * ctx[b * ENC_ + k];
        out_stop[b * T_ + tm] = acc;
    }
}

// ---------------- setup kernel: pack LSTM weights to bf16 WMMA-B layout ----------------
__global__ __launch_bounds__(TPB_) void pack_weights_kernel(
    const float* __restrict__ w_ih1, const float* __restrict__ w_hh1,
    const float* __restrict__ w_ih2, const float* __restrict__ w_hh2,
    __bf16* __restrict__ W1p, __bf16* __restrict__ W2p)
{
    const int id = blockIdx.x * TPB_ + threadIdx.x;
    const int n1 = NTILES * KT1 * 32;
    const int n2 = NTILES * KT2 * 32;
    if (id < n1) {
        const int lane  = id & 31;
        const int kt    = (id >> 5) % KT1;
        const int ntile = (id >> 5) / KT1;
        const int N  = ntile * 16 + (lane & 15);
        const int kb = ((lane >> 4) & 1) << 4;
        __bf16* dst = W1p + (size_t)id * 16;
        for (int i = 0; i < 16; ++i) {
            const int K = kt * 32 + kb + i;
            const float v = (K < PRE_ + ENC_) ? w_ih1[N * (PRE_ + ENC_) + K]
                                              : w_hh1[N * ARNN_ + (K - (PRE_ + ENC_))];
            dst[i] = f2bf(v);
        }
    } else if (id < n1 + n2) {
        const int id2   = id - n1;
        const int lane  = id2 & 31;
        const int kt    = (id2 >> 5) % KT2;
        const int ntile = (id2 >> 5) / KT2;
        const int N  = ntile * 16 + (lane & 15);
        const int kb = ((lane >> 4) & 1) << 4;
        __bf16* dst = W2p + (size_t)id2 * 16;
        for (int i = 0; i < 16; ++i) {
            const int K = kt * 32 + kb + i;
            const float v = (K < ARNN_ + ENC_) ? w_ih2[N * (ARNN_ + ENC_) + K]
                                               : w_hh2[N * DRNN_ + (K - (ARNN_ + ENC_))];
            dst[i] = f2bf(v);
        }
    }
}

// ---------------- setup kernel: prenet over whole teacher-forced sequence ----------------
__global__ __launch_bounds__(TPB_) void prenet_kernel(
    const float* __restrict__ mels, const float* __restrict__ w1,
    const float* __restrict__ w2, float* __restrict__ xs)
{
    __shared__ float h1[PRE_];
    const int row = blockIdx.x;       // row = t*B + b, t in [0,T)
    const int t = row / B_, b = row % B_;
    const int d = threadIdx.x;
    float acc = 0.f;
    if (t > 0) {
        const float* mr = mels + ((size_t)b * T_ + (t - 1)) * NM_;
        const float* wr = w1 + d * NM_;
        for (int k = 0; k < NM_; ++k) acc += wr[k] * mr[k];
    }
    h1[d] = fmaxf(acc, 0.f);
    __syncthreads();
    float acc2 = 0.f;
    const float* wr2 = w2 + d * PRE_;
    for (int k = 0; k < PRE_; ++k) acc2 += wr2[k] * h1[k];
    xs[(size_t)row * PRE_ + d] = fmaxf(acc2, 0.f);
}

// ---------------- setup kernel: enc_cache = enc @ w_enc^T ----------------
__global__ __launch_bounds__(TPB_) void enccache_kernel(
    const float* __restrict__ enc, const float* __restrict__ w_enc,
    float* __restrict__ encc)
{
    const int id = blockIdx.x * TPB_ + threadIdx.x;   // B*S*AD = 2,097,152
    if (id >= B_ * S_ * AD_) return;
    const int d  = id & (AD_ - 1);
    const int bs = id >> 7;
    const float* er = enc + (size_t)bs * ENC_;
    const float* wr = w_enc + d * ENC_;
    float acc = 0.f;
    for (int k = 0; k < ENC_; ++k) acc += wr[k] * er[k];
    encc[id] = acc;
}

// ---------------- persistent decoder kernel ----------------
__global__ __launch_bounds__(TPB_) void decoder_persistent(
    const float* __restrict__ xs, const float* __restrict__ encoder_outputs,
    const unsigned char* __restrict__ mask,
    const float* __restrict__ encc, float* __restrict__ locp,
    const v16bf* __restrict__ W1p, const v16bf* __restrict__ W2p,
    const float* __restrict__ b_ih1, const float* __restrict__ b_hh1,
    const float* __restrict__ w_q, const float* __restrict__ b_q,
    const float* __restrict__ k_loc, const float* __restrict__ w_loc,
    const float* __restrict__ w_e,
    const float* __restrict__ b_ih2, const float* __restrict__ b_hh2,
    const float* __restrict__ w_mel, const float* __restrict__ b_mel,
    const float* __restrict__ w_stop, const float* __restrict__ b_stop,
    float* __restrict__ ah0, float* __restrict__ ah1, float* __restrict__ ac,
    float* __restrict__ dh0, float* __restrict__ dh1, float* __restrict__ dc,
    float* __restrict__ ctx, float* __restrict__ aw, float* __restrict__ cum,
    unsigned* __restrict__ bar,
    float* __restrict__ out_mel, float* __restrict__ out_stop, float* __restrict__ out_attn)
{
    const int tid  = threadIdx.x;
    const int lane = tid & 31;
    const int wid  = tid >> 5;
    const int gw   = blockIdx.x * (TPB_ / 32) + wid;   // global wave id, 0..511

    __shared__ __align__(32) __bf16 alds[2 * AFRAG_];  // 32 KB A-staging, double buffered
    __shared__ float sq[AD_];
    __shared__ float se[S_];
    __shared__ float sr[TPB_];

    unsigned bars = 0;

    for (int t = 0; t < T_; ++t) {
        const float* ahp = (t & 1) ? ah1 : ah0;
        float*       ahn = (t & 1) ? ah0 : ah1;
        const float* dhp = (t & 1) ? dh1 : dh0;
        float*       dhn = (t & 1) ? dh0 : dh1;

        // ============ window A: attention LSTM + location conv/proj + prev mel ============
        if (blockIdx.x < 8) {
            // A1 = [xs_t | ctx | ah_prev], k-tile boundaries 8 and 24
            lstm_gates_gemm_block(wid, blockIdx.x * 8, lane, W1p, KT1,
                                  xs + (size_t)t * B_ * PRE_, PRE_, 8,
                                  ctx, ENC_, 24,
                                  ahp, ARNN_,
                                  b_ih1, b_hh1, ac, ahn, alds);
        } else if (gw < 448) {
            // location conv (SAME, K=31, 2 in-ch) fused with NF->AD projection
            for (int u = gw - 64; u < B_ * 16; u += 384) {
                const int b = u >> 4;
                const int s = ((u & 15) << 5) + lane;
                float accf[NF_];
#pragma unroll
                for (int f = 0; f < NF_; ++f) accf[f] = 0.f;
                for (int k = 0; k < KS_; ++k) {
                    const int s2 = s + k - 15;
                    float a0 = 0.f, a1 = 0.f;
                    if (s2 >= 0 && s2 < S_) {
                        a0 = aw[b * S_ + s2];
                        a1 = cum[b * S_ + s2];
                    }
#pragma unroll
                    for (int f = 0; f < NF_; ++f)
                        accf[f] += k_loc[(f * 2 + 0) * KS_ + k] * a0 +
                                   k_loc[(f * 2 + 1) * KS_ + k] * a1;
                }
                float* lp = locp + ((size_t)b * S_ + s) * AD_;
                for (int d = 0; d < AD_; ++d) {
                    float v = 0.f;
#pragma unroll
                    for (int f = 0; f < NF_; ++f) v += w_loc[d * NF_ + f] * accf[f];
                    lp[d] = v;
                }
            }
        } else {
            // mel / stop projection for the previous step (uses dh_{t-1}, ctx_{t-1})
            if (t > 0) {
                for (int o = (gw - 448) * 32 + lane; o < B_ * NM_ + B_; o += 64 * 32)
                    mel_stop_out(o, t - 1, dhp, ctx, w_mel, b_mel, w_stop, b_stop,
                                 out_mel, out_stop);
            }
        }
        ++bars; grid_barrier(bar, bars * GRID_);

        // ============ window B: location-sensitive attention (WG b handles batch b) ============
        if (blockIdx.x < B_) {
            const int b = blockIdx.x;
            if (tid < AD_) {
                float acc = b_q[tid];
                const float* wr = w_q + tid * ARNN_;
                const float* hr = ahn + b * ARNN_;
                for (int k = 0; k < ARNN_; ++k) acc += wr[k] * hr[k];
                sq[tid] = acc;
            }
            __syncthreads();
            for (int s = tid; s < S_; s += TPB_) {
                const float* er = encc + ((size_t)b * S_ + s) * AD_;
                const float* lr = locp + ((size_t)b * S_ + s) * AD_;
                float acc = 0.f;
                for (int d = 0; d < AD_; ++d) acc += w_e[d] * tanhf(sq[d] + er[d] + lr[d]);
                if (mask[b * S_ + s]) acc -= 1e9f;
                se[s] = acc;
            }
            __syncthreads();
            // softmax over S
            float mx = -3.4e38f;
            for (int s = tid; s < S_; s += TPB_) mx = fmaxf(mx, se[s]);
            sr[tid] = mx; __syncthreads();
            for (int st = TPB_ / 2; st > 0; st >>= 1) {
                if (tid < st) sr[tid] = fmaxf(sr[tid], sr[tid + st]);
                __syncthreads();
            }
            mx = sr[0]; __syncthreads();
            float ls = 0.f;
            for (int s = tid; s < S_; s += TPB_) {
                const float e = expf(se[s] - mx);
                se[s] = e; ls += e;
            }
            sr[tid] = ls; __syncthreads();
            for (int st = TPB_ / 2; st > 0; st >>= 1) {
                if (tid < st) sr[tid] += sr[tid + st];
                __syncthreads();
            }
            const float inv = 1.f / sr[0];
            __syncthreads();
            for (int s = tid; s < S_; s += TPB_) {
                const float a = se[s] * inv;
                se[s] = a;
                aw[b * S_ + s]  = a;
                cum[b * S_ + s] += a;
                out_attn[((size_t)b * T_ + t) * S_ + s] = a;
            }
            __syncthreads();
            // ctx = aw @ enc
            for (int d = tid; d < ENC_; d += TPB_) {
                float acc = 0.f;
                const float* ep = encoder_outputs + (size_t)b * S_ * ENC_ + d;
                for (int s = 0; s < S_; ++s) acc += se[s] * ep[(size_t)s * ENC_];
                ctx[b * ENC_ + d] = acc;
            }
        }
        ++bars; grid_barrier(bar, bars * GRID_);

        // ============ window C: decoder LSTM ============
        if (blockIdx.x < 8) {
            // A2 = [ah_t | ctx | dh_prev], k-tile boundaries 32 and 48
            lstm_gates_gemm_block(wid, blockIdx.x * 8, lane, W2p, KT2,
                                  ahn, ARNN_, 32,
                                  ctx, ENC_, 48,
                                  dhp, DRNN_,
                                  b_ih2, b_hh2, dc, dhn, alds);
        }
        ++bars; grid_barrier(bar, bars * GRID_);
    }

    // final mel/stop for t = T-1  (dh written at t=599 is dh0 since T is even)
    {
        const float* dhf = (T_ & 1) ? dh1 : dh0;
        for (int o = blockIdx.x * TPB_ + tid; o < B_ * NM_ + B_; o += GRID_ * TPB_)
            mel_stop_out(o, T_ - 1, dhf, ctx, w_mel, b_mel, w_stop, b_stop,
                         out_mel, out_stop);
    }
}

// ---------------- host side ----------------
extern "C" void kernel_launch(void* const* d_in, const int* in_sizes, int n_in,
                              void* d_out, int out_size, void* d_ws, size_t ws_size,
                              hipStream_t stream)
{
    (void)in_sizes; (void)n_in; (void)out_size; (void)ws_size;

    const float*         encoder_outputs = (const float*)d_in[0];
    const unsigned char* encoder_mask    = (const unsigned char*)d_in[1];
    const float*         mels            = (const float*)d_in[2];
    const float* w_pre1 = (const float*)d_in[3];
    const float* w_pre2 = (const float*)d_in[4];
    const float* w_ih1  = (const float*)d_in[5];
    const float* w_hh1  = (const float*)d_in[6];
    const float* b_ih1  = (const float*)d_in[7];
    const float* b_hh1  = (const float*)d_in[8];
    const float* w_q    = (const float*)d_in[9];
    const float* b_q    = (const float*)d_in[10];
    const float* w_enc  = (const float*)d_in[11];
    const float* k_loc  = (const float*)d_in[12];
    const float* w_loc  = (const float*)d_in[13];
    const float* w_e    = (const float*)d_in[14];
    const float* w_ih2  = (const float*)d_in[15];
    const float* w_hh2  = (const float*)d_in[16];
    const float* b_ih2  = (const float*)d_in[17];
    const float* b_hh2  = (const float*)d_in[18];
    const float* w_mel  = (const float*)d_in[19];
    const float* b_mel  = (const float*)d_in[20];
    const float* w_stop = (const float*)d_in[21];
    const float* b_stop = (const float*)d_in[22];

    float* out_mel  = (float*)d_out;
    float* out_stop = out_mel + (size_t)B_ * T_ * NM_;
    float* out_attn = out_stop + (size_t)B_ * T_;

    // workspace carve-out (256B aligned chunks)
    char* ws = (char*)d_ws;
    size_t off = 0;
    auto carve = [&](size_t bytes) -> void* {
        void* p = ws + off;
        off = (off + bytes + 255) & ~(size_t)255;
        return p;
    };
    __bf16* W1p  = (__bf16*)carve((size_t)NGATE * K1DIM * 2);      // 14.68 MB
    __bf16* W2p  = (__bf16*)carve((size_t)NGATE * K2DIM * 2);      // 20.97 MB
    float*  xs   = (float*)carve((size_t)T_ * B_ * PRE_ * 4);      // 19.66 MB
    float*  encc = (float*)carve((size_t)B_ * S_ * AD_ * 4);       // 8.39 MB
    float*  locp = (float*)carve((size_t)B_ * S_ * AD_ * 4);       // 8.39 MB

    char* state_begin = ws + off;
    float* ah0 = (float*)carve((size_t)B_ * ARNN_ * 4);
    float* ah1 = (float*)carve((size_t)B_ * ARNN_ * 4);
    float* ac  = (float*)carve((size_t)B_ * ARNN_ * 4);
    float* dh0 = (float*)carve((size_t)B_ * DRNN_ * 4);
    float* dh1 = (float*)carve((size_t)B_ * DRNN_ * 4);
    float* dc  = (float*)carve((size_t)B_ * DRNN_ * 4);
    float* ctx = (float*)carve((size_t)B_ * ENC_ * 4);
    float* aw  = (float*)carve((size_t)B_ * S_ * 4);
    float* cum = (float*)carve((size_t)B_ * S_ * 4);
    unsigned* bar = (unsigned*)carve(256);
    size_t state_bytes = (size_t)((ws + off) - state_begin);

    // zero all recurrent state + barrier counter (graph-capturable)
    hipMemsetAsync(state_begin, 0, state_bytes, stream);

    // setup: pack weights, prenet, encoder cache
    {
        const int total = NTILES * KT1 * 32 + NTILES * KT2 * 32;   // 1,114,112
        pack_weights_kernel<<<(total + TPB_ - 1) / TPB_, TPB_, 0, stream>>>(
            w_ih1, w_hh1, w_ih2, w_hh2, W1p, W2p);
    }
    prenet_kernel<<<T_ * B_, TPB_, 0, stream>>>(mels, w_pre1, w_pre2, xs);
    enccache_kernel<<<(B_ * S_ * AD_ + TPB_ - 1) / TPB_, TPB_, 0, stream>>>(
        encoder_outputs, w_enc, encc);

    // persistent decoder loop
    decoder_persistent<<<GRID_, TPB_, 0, stream>>>(
        xs, encoder_outputs, encoder_mask, encc, locp,
        (const v16bf*)W1p, (const v16bf*)W2p,
        b_ih1, b_hh1, w_q, b_q, k_loc, w_loc, w_e,
        b_ih2, b_hh2, w_mel, b_mel, w_stop, b_stop,
        ah0, ah1, ac, dh0, dh1, dc, ctx, aw, cum, bar,
        out_mel, out_stop, out_attn);
}
